// UnifiedGraphQLSTM_65481071406917
// MI455X (gfx1250) — matrix operand order repaired and checked by reading, not compiled, not run on hardware
//
#include <hip/hip_runtime.h>
#include <hip/hip_bf16.h>
#include <cstdint>

// ---------------- problem constants ----------------
constexpr int T    = 256;
constexpr int B    = 256;
constexpr int F    = 128;
constexpr int H1   = 256;
constexpr int H    = 128;
constexpr int TAGS = 32;

// ---------------- feature probes ----------------
#if defined(__has_builtin)
#if __has_builtin(__builtin_amdgcn_global_load_async_to_lds_b128)
#define HAVE_ASYNC_LDS 1
#endif
#if __has_builtin(__builtin_amdgcn_s_wait_asynccnt)
#define HAVE_WAIT_ASYNC 1
#endif
#endif

// ---------------- WMMA types / helpers ----------------
typedef __attribute__((ext_vector_type(16))) __bf16 v16bf;
typedef __attribute__((ext_vector_type(8)))  float  v8f;
typedef __attribute__((ext_vector_type(4)))  int    v4i;

typedef __attribute__((address_space(1))) v4i* gptr_v4i;   // global int4*
typedef __attribute__((address_space(3))) v4i* lptr_v4i;   // LDS int4*

__device__ __forceinline__ v8f wmma_bf(v16bf a, v16bf b, v8f c) {
    // D = A(16x32 bf16) x B(32x16 bf16) + C(16x16 f32)
    return __builtin_amdgcn_wmma_f32_16x16x32_bf16(
        /*neg_a=*/false, a, /*neg_b=*/false, b,
        /*c_mod=*/(short)0, c, /*reuse_a=*/false, /*reuse_b=*/false);
}

// A-fragment (16x32 bf16) from row-major bf16 row pointer.
// half=lane/16; VGPR j(0..7): K = kbase+16*half+2j
__device__ __forceinline__ v16bf load_a(const __bf16* rowp, int kbase, int half) {
    union { uint32_t u[8]; v16bf v; } a;
    const uint32_t* p0 = (const uint32_t*)(rowp + kbase + 16 * half);
    const uint32_t* p1 = (const uint32_t*)(rowp + kbase + 16 * half + 8);
#pragma unroll
    for (int j = 0; j < 4; ++j) { a.u[j] = p0[j]; a.u[4 + j] = p1[j]; }
    return a.v;
}

// B-fragment from pre-swizzled storage: 8 consecutive dwords per lane.
__device__ __forceinline__ v16bf load_b(const uint32_t* p) {
    union { uint32_t u[8]; v16bf v; } b;
#pragma unroll
    for (int j = 0; j < 8; ++j) b.u[j] = p[j];
    return b.v;
}

__device__ __forceinline__ uint32_t packbf(float lo, float hi) {
    union { __bf16 b[2]; uint32_t u; } p;
    p.b[0] = (__bf16)lo; p.b[1] = (__bf16)hi;
    return p.u;
}

// Branchless fast transcendental helpers (v_exp_f32 / v_rcp_f32; no EXEC churn
// between WMMAs, correct saturation: exp->inf => rcp->0 => tanh->sign(x)*1).
__device__ __forceinline__ float fast_tanh(float x) {
    float ax = __builtin_fabsf(x);
    float e  = __expf(ax + ax);
    float r  = 1.0f - 2.0f * __builtin_amdgcn_rcpf(e + 1.0f);
    return __builtin_copysignf(r, x);
}
__device__ __forceinline__ float fast_sig(float x) {
    return __builtin_amdgcn_rcpf(1.0f + __expf(-x));
}

// ---------------- workspace layout (bytes) ----------------
constexpr size_t O_W1S   = 0;                       // 16384 dwords  (64 KB)
constexpr size_t O_W2S   = 65536;                   // 16384 dwords  (64 KB)
constexpr size_t O_ADJB  = 131072;                  // 65536 bf16    (128 KB)
constexpr size_t O_WCATS = 262144;                  // 2048 dwords   (8 KB)
constexpr size_t O_WTAGS = 270336;                  // 2048 dwords   (8 KB)
constexpr size_t O_BCAT  = 278528;                  // 16 f32
constexpr size_t O_NSSWZ = 524288;                  // 16 MB  node_states, B-frag swizzled bf16
constexpr size_t O_AGGB  = O_NSSWZ + 16777216;      // 16 MB  aggregated bf16 row-major [T,B,H]
constexpr size_t O_LSTMB = O_AGGB  + 16777216;      // 16 MB  lstm_out bf16 row-major [T*B,H]
constexpr size_t O_LOGIT = O_LSTMB + 16777216;      //  8 MB  logits f32 [T*B,TAGS]

// ---------------- kernel 0: weight prep (f32 -> bf16, B-fragment swizzle) ----------------
__global__ void prep_kernel(const float* __restrict__ W1, const float* __restrict__ W2,
                            const float* __restrict__ adj, const float* __restrict__ Win,
                            const float* __restrict__ b_in, const float* __restrict__ Wtag,
                            uint32_t* w1s, uint32_t* w2s, __bf16* adjb,
                            uint32_t* wcats, float* bcat, uint32_t* wtags) {
    int tid = blockIdx.x * blockDim.x + threadIdx.x;
    int nth = gridDim.x * blockDim.x;

    // W1 [128,256] -> s = nt*1024 + kt*256 + lane*8 + j  (nt<16, kt<4)
    for (int s = tid; s < 16384; s += nth) {
        int j = s & 7, lane = (s >> 3) & 31, kt = (s >> 8) & 3, nt = s >> 10;
        int k = 32 * kt + 16 * (lane >> 4) + 2 * j;
        int n = 16 * nt + (lane & 15);
        w1s[s] = packbf(W1[k * H1 + n], W1[(k + 1) * H1 + n]);
    }
    // W2 [256,128] -> s = nt*2048 + kt*256 + lane*8 + j  (nt<8, kt<8)
    for (int s = tid; s < 16384; s += nth) {
        int j = s & 7, lane = (s >> 3) & 31, kt = (s >> 8) & 7, nt = s >> 11;
        int k = 32 * kt + 16 * (lane >> 4) + 2 * j;
        int n = 16 * nt + (lane & 15);
        w2s[s] = packbf(W2[k * H + n], W2[(k + 1) * H + n]);
    }
    // adj -> plain bf16 (A operand, row-major)
    for (int s = tid; s < B * B; s += nth) adjb[s] = (__bf16)adj[s];
    // Wcat [256,16]: col n = 4g+q from Win[g,k,q]; s = kt*256 + lane*8 + j (kt<8)
    for (int s = tid; s < 2048; s += nth) {
        int j = s & 7, lane = (s >> 3) & 31, kt = s >> 8;
        int k = 32 * kt + 16 * (lane >> 4) + 2 * j;
        int n = lane & 15, g = n >> 2, q = n & 3;
        float lo = Win[((g * 256) + k) * 4 + q];
        float hi = Win[((g * 256) + k + 1) * 4 + q];
        wcats[s] = packbf(lo, hi);
    }
    for (int s = tid; s < 16; s += nth) bcat[s] = b_in[s];
    // Wtag [128,32] -> s = nt*1024 + kt*256 + lane*8 + j (nt<2, kt<4)
    for (int s = tid; s < 2048; s += nth) {
        int j = s & 7, lane = (s >> 3) & 31, kt = (s >> 8) & 3, nt = s >> 10;
        int k = 32 * kt + 16 * (lane >> 4) + 2 * j;
        int n = 16 * nt + (lane & 15);
        wtags[s] = packbf(Wtag[k * TAGS + n], Wtag[(k + 1) * TAGS + n]);
    }
}

// ---------------- kernel 1: fused 2-layer MLP (tanh) ----------------
__global__ __launch_bounds__(128) void mlp_kernel(const float* __restrict__ X,
                                                  const float* __restrict__ b1,
                                                  const float* __restrict__ b2,
                                                  const uint32_t* __restrict__ w1s,
                                                  const uint32_t* __restrict__ w2s,
                                                  __bf16* __restrict__ nsswz) {
    __shared__ __bf16 Xb[64 * 128];        // 16 KB input tile (bf16)
    __shared__ __bf16 A1[4][16 * 256];     // 32 KB hidden activations per wave

    int tid = threadIdx.x;
    int rowBase = blockIdx.x * 64;

    const float4* Xv = (const float4*)(X + (size_t)rowBase * F);
    for (int i = tid; i < 64 * 128 / 4; i += 128) {
        float4 x = Xv[i];
        int b0 = i * 4;
        Xb[b0 + 0] = (__bf16)x.x; Xb[b0 + 1] = (__bf16)x.y;
        Xb[b0 + 2] = (__bf16)x.z; Xb[b0 + 3] = (__bf16)x.w;
    }
    __syncthreads();

    int wave = tid >> 5, lane = tid & 31, l15 = lane & 15, half = lane >> 4;
    const __bf16* arow = &Xb[(wave * 16 + l15) * F];
    __bf16* a1w = A1[wave];

    // layer 1: [16,128] @ [128,256] -> tanh -> LDS bf16
    for (int nt = 0; nt < 16; ++nt) {
        v8f acc = {};
#pragma unroll
        for (int kt = 0; kt < 4; ++kt) {
            v16bf av = load_a(arow, 32 * kt, half);
            v16bf bv = load_b(w1s + ((nt * 4 + kt) * 32 + lane) * 8);
            acc = wmma_bf(av, bv, acc);
        }
        int col = nt * 16 + l15;
        float bias = b1[col];
#pragma unroll
        for (int r = 0; r < 8; ++r) {
            int m = r + 8 * half;
            a1w[m * H1 + col] = (__bf16)fast_tanh(acc[r] + bias);
        }
    }

    // layer 2: [16,256] @ [256,128] -> tanh -> global (swizzled B-frag layout)
    const __bf16* a2row = &a1w[l15 * H1];
    int tblk  = rowBase >> 8;                 // block-constant t
    int jbase = (rowBase & 255) + wave * 16;  // wave-constant j base
    for (int nt = 0; nt < 8; ++nt) {
        v8f acc = {};
#pragma unroll
        for (int kt = 0; kt < 8; ++kt) {
            v16bf av = load_a(a2row, 32 * kt, half);
            v16bf bv = load_b(w2s + ((nt * 8 + kt) * 32 + lane) * 8);
            acc = wmma_bf(av, bv, acc);
        }
        int col = nt * 16 + l15;
        float bias = b2[col];
        size_t ntBase = ((size_t)tblk * 8 + nt) * 8;   // dword-tile base for (t, nt)
#pragma unroll
        for (int r = 0; r < 8; ++r) {
            int m = r + 8 * half;
            float v = fast_tanh(acc[r] + bias);
            int jrow = jbase + m;
            int kt2 = jrow >> 5, r32 = jrow & 31;
            int hh = r32 >> 4, k2 = r32 & 15;
            size_t dw = ((ntBase + kt2) * 32 + (hh * 16 + l15)) * 8 + (k2 >> 1);
            nsswz[dw * 2 + (k2 & 1)] = (__bf16)v;
        }
    }
}

// ---------------- kernel 2: aggregation  out[t] = adj @ ns[t] ----------------
__global__ __launch_bounds__(128) void agg_kernel(const __bf16* __restrict__ adjb,
                                                  const __bf16* __restrict__ nsswz,
                                                  __bf16* __restrict__ aggb) {
    int wave = threadIdx.x >> 5, lane = threadIdx.x & 31, l15 = lane & 15, half = lane >> 4;
    int tileId = blockIdx.x * 4 + wave;          // < 256*16*8 = 32768
    int t = tileId >> 7;
    int rem = tileId & 127;
    int it = rem >> 3, ht = rem & 7;

    const __bf16* arow = adjb + (it * 16 + l15) * B;
    const uint32_t* bbase = (const uint32_t*)nsswz + ((size_t)t * 8 + ht) * 8 * 256;

    v8f acc = {};
#pragma unroll
    for (int kt = 0; kt < 8; ++kt) {
        v16bf av = load_a(arow, 32 * kt, half);
        v16bf bv = load_b(bbase + kt * 256 + lane * 8);
        acc = wmma_bf(av, bv, acc);
    }
    int col = ht * 16 + l15;
#pragma unroll
    for (int r = 0; r < 8; ++r) {
        int m = r + 8 * half;
        aggb[((size_t)t * B + it * 16 + m) * H + col] = (__bf16)acc[r];
    }
}

// ---------------- kernel 3: persistent QLSTM scan ----------------
// One 1024-thread block. h (bf16) in LDS, c (f32) in registers. Wcat B-fragments
// staged into LDS once via CDNA5 async global->LDS DMA, then reused 256 steps.
__global__ __launch_bounds__(1024) void lstm_kernel(const __bf16* __restrict__ aggb,
                                                    const uint32_t* __restrict__ wcats,
                                                    const float* __restrict__ bcat,
                                                    const float* __restrict__ Wout,
                                                    const float* __restrict__ b_out,
                                                    __bf16* __restrict__ lstmb) {
    __shared__ __bf16   hbuf[B * H];       // 64 KB
    __shared__ float    zbuf[B * 16];      // 16 KB
    __shared__ uint32_t wcatS[8 * 256];    //  8 KB swizzled Wcat B-fragments
    __shared__ float    WoutS[16 * H];     //  8 KB
    __shared__ float    boutS[4 * H];      //  2 KB
    __shared__ float    bcatS[16];

    int tid = threadIdx.x;

    // ---- stage Wcat into LDS (async DMA path if available) ----
#if defined(HAVE_ASYNC_LDS)
    if (tid < 512) {
        gptr_v4i gsrc = (gptr_v4i)((const char*)wcats + tid * 16);
        lptr_v4i ldst = (lptr_v4i)(&wcatS[tid * 4]);
        __builtin_amdgcn_global_load_async_to_lds_b128(gsrc, ldst, 0, 0);
    }
#if defined(HAVE_WAIT_ASYNC)
    __builtin_amdgcn_s_wait_asynccnt(0);
#else
    asm volatile("s_wait_asynccnt 0" ::: "memory");
#endif
#else
    for (int i = tid; i < 8 * 256; i += 1024) wcatS[i] = wcats[i];
#endif

    for (int i = tid; i < B * H; i += 1024) hbuf[i] = (__bf16)0.0f;
    for (int i = tid; i < 16 * H; i += 1024) WoutS[i] = Wout[i];
    for (int i = tid; i < 4 * H; i += 1024) boutS[i] = b_out[i];
    if (tid < 16) bcatS[tid] = bcat[tid];

    float c[32];
#pragma unroll
    for (int i = 0; i < 32; ++i) c[i] = 0.0f;
    __syncthreads();

    int wave = tid >> 5, lane = tid & 31, l15 = lane & 15, half = lane >> 4;

    for (int t = 0; t < T; ++t) {
        if (wave < 16) {
            const __bf16* xrow = aggb + ((size_t)t * B + wave * 16 + l15) * H;
            const __bf16* hrow = &hbuf[(wave * 16 + l15) * H];
            v8f acc = {};
#pragma unroll
            for (int kt = 0; kt < 8; ++kt) {
                v16bf av = (kt < 4) ? load_a(xrow, 32 * kt, half)
                                    : load_a(hrow, 32 * (kt - 4), half);
                v16bf bv = load_b(&wcatS[kt * 256 + lane * 8]);
                acc = wmma_bf(av, bv, acc);
            }
            float bias = bcatS[l15];
#pragma unroll
            for (int r = 0; r < 8; ++r) {
                int m = r + 8 * half;
                zbuf[(wave * 16 + m) * 16 + l15] = fast_tanh(acc[r] + bias);
            }
        }
        __syncthreads();

#pragma unroll 4
        for (int idx = 0; idx < 32; ++idx) {
            int lin = idx * 1024 + tid;
            int bb = lin >> 7, hc = lin & 127;
            const float* zr = &zbuf[bb * 16];
            float g0 = boutS[0 * H + hc], g1 = boutS[1 * H + hc];
            float g2 = boutS[2 * H + hc], g3 = boutS[3 * H + hc];
#pragma unroll
            for (int q = 0; q < 4; ++q) {
                g0 += zr[0 + q]  * WoutS[(0 + q)  * H + hc];
                g1 += zr[4 + q]  * WoutS[(4 + q)  * H + hc];
                g2 += zr[8 + q]  * WoutS[(8 + q)  * H + hc];
                g3 += zr[12 + q] * WoutS[(12 + q) * H + hc];
            }
            float fg = fast_sig(g0), ig = fast_sig(g1);
            float gg = fast_tanh(g2), og = fast_sig(g3);
            float cn = fg * c[idx] + ig * gg;
            c[idx] = cn;
            float hv = og * fast_tanh(cn);
            lstmb[(size_t)t * B * H + lin] = (__bf16)hv;
            hbuf[lin] = (__bf16)hv;
        }
        __syncthreads();
    }
}

// ---------------- kernel 4: tag logits ----------------
__global__ __launch_bounds__(128) void logits_kernel(const __bf16* __restrict__ lstmb,
                                                     const uint32_t* __restrict__ wtags,
                                                     const float* __restrict__ btag,
                                                     float* __restrict__ logits) {
    int wave = threadIdx.x >> 5, lane = threadIdx.x & 31, l15 = lane & 15, half = lane >> 4;
    int rt = blockIdx.x * 4 + wave;              // < 4096 row tiles
    const __bf16* arow = lstmb + ((size_t)rt * 16 + l15) * H;
#pragma unroll
    for (int nt = 0; nt < 2; ++nt) {
        v8f acc = {};
#pragma unroll
        for (int kt = 0; kt < 4; ++kt) {
            v16bf av = load_a(arow, 32 * kt, half);
            v16bf bv = load_b(wtags + ((nt * 4 + kt) * 32 + lane) * 8);
            acc = wmma_bf(av, bv, acc);
        }
        int col = nt * 16 + l15;
        float bias = btag[col];
#pragma unroll
        for (int r = 0; r < 8; ++r) {
            int m = r + 8 * half;
            logits[((size_t)rt * 16 + m) * TAGS + col] = acc[r] + bias;
        }
    }
}

// ---------------- kernel 5: row log-softmax over 32 tags ----------------
__global__ void lsm_kernel(const float* __restrict__ logits, float* __restrict__ out) {
    int row = blockIdx.x * blockDim.x + threadIdx.x;     // 65536 rows
    const float* x = logits + (size_t)row * TAGS;
    float mx = -3.4e38f;
#pragma unroll
    for (int i = 0; i < TAGS; ++i) mx = fmaxf(mx, x[i]);
    float s = 0.0f;
#pragma unroll
    for (int i = 0; i < TAGS; ++i) s += __expf(x[i] - mx);
    float lse = mx + __logf(s);
    float* o = out + (size_t)row * TAGS;
#pragma unroll
    for (int i = 0; i < TAGS; ++i) o[i] = x[i] - lse;
}

// ---------------- launcher ----------------
extern "C" void kernel_launch(void* const* d_in, const int* in_sizes, int n_in,
                              void* d_out, int out_size, void* d_ws, size_t ws_size,
                              hipStream_t stream) {
    (void)in_sizes; (void)n_in; (void)out_size; (void)ws_size;

    const float* X     = (const float*)d_in[0];   // [T,B,F]
    const float* adj   = (const float*)d_in[1];   // [B,B]
    const float* W1    = (const float*)d_in[2];   // [F,H1]
    const float* b1    = (const float*)d_in[3];
    const float* W2    = (const float*)d_in[4];   // [H1,H]
    const float* b2    = (const float*)d_in[5];
    const float* Win   = (const float*)d_in[6];   // [4,F+H,Q]
    const float* b_in  = (const float*)d_in[7];   // [4,Q]
    const float* Wout  = (const float*)d_in[8];   // [4,Q,H]
    const float* b_out = (const float*)d_in[9];   // [4,H]
    const float* Wtag  = (const float*)d_in[10];  // [H,TAGS]
    const float* btag  = (const float*)d_in[11];  // [TAGS]
    float* out = (float*)d_out;

    char* ws = (char*)d_ws;
    uint32_t* w1s   = (uint32_t*)(ws + O_W1S);
    uint32_t* w2s   = (uint32_t*)(ws + O_W2S);
    __bf16*   adjb  = (__bf16*)  (ws + O_ADJB);
    uint32_t* wcats = (uint32_t*)(ws + O_WCATS);
    uint32_t* wtags = (uint32_t*)(ws + O_WTAGS);
    float*    bcat  = (float*)   (ws + O_BCAT);
    __bf16*   nsswz = (__bf16*)  (ws + O_NSSWZ);
    __bf16*   aggb  = (__bf16*)  (ws + O_AGGB);
    __bf16*   lstmb = (__bf16*)  (ws + O_LSTMB);
    float*    logit = (float*)   (ws + O_LOGIT);

    prep_kernel<<<256, 256, 0, stream>>>(W1, W2, adj, Win, b_in, Wtag,
                                         w1s, w2s, adjb, wcats, bcat, wtags);
    mlp_kernel<<<(T * B) / 64, 128, 0, stream>>>(X, b1, b2, w1s, w2s, nsswz);
    agg_kernel<<<(T * 16 * 8) / 4, 128, 0, stream>>>(adjb, nsswz, aggb);
    lstm_kernel<<<1, 1024, 0, stream>>>(aggb, wcats, bcat, Wout, b_out, lstmb);
    logits_kernel<<<(T * B / 16) / 4, 128, 0, stream>>>(lstmb, wtags, btag, logit);
    lsm_kernel<<<256, 256, 0, stream>>>(logit, out);
}
